// LlamaDecoderLayer_18382460027076
// MI455X (gfx1250) — compile-verified, matrix operand order checked by reading, not run on hardware
//
#include <hip/hip_runtime.h>

// ---------------------------------------------------------------------------
// LlamaDecoderLayer for MI455X (gfx1250): bf16 WMMA GEMMs + flash attention.
// B=4, S=2048, H=576, NH=9, HD=64, IM=1536.  All GEMM dims divide the tiles.
// GEMM weight tiles are staged to LDS via the CDNA5 async global->LDS path
// (ASYNCcnt), double-buffered to overlap copy with WMMA compute.
// ---------------------------------------------------------------------------

typedef __bf16 bf16;
typedef __attribute__((ext_vector_type(8)))  __bf16 v8bf;
typedef __attribute__((ext_vector_type(16))) __bf16 v16bf;
typedef __attribute__((ext_vector_type(8)))  float  v8f;

#define Bv 4
#define Sv 2048
#define Hv 576
#define NHv 9
#define HDv 64
#define IMv 1536
#define Mtok (Bv * Sv)          // 8192 token rows

// Load one 16-element bf16 WMMA operand fragment (A-style / B-on-W-rows style).
// base = row pointer + k-tile offset; half = lane>>4.
// elem i: k = (i<8 ? 0 : 16) + 8*half + (i%8)   (ISA 7.12.2, 16-bit A 16x32)
__device__ __forceinline__ v16bf load_frag16(const bf16* base, int half) {
  v8bf lo = *(const v8bf*)(base + 8 * half);
  v8bf hi = *(const v8bf*)(base + 16 + 8 * half);
  return __builtin_shufflevector(lo, hi, 0, 1, 2, 3, 4, 5, 6, 7,
                                         8, 9, 10, 11, 12, 13, 14, 15);
}

// 16-byte global -> LDS copy via CDNA5 async path (ASYNCcnt tracked).
#if defined(__AMDGCN__)
typedef int v4i_g __attribute__((vector_size(16)));   // matches builtin's V4i
#endif
__device__ __forceinline__ void async_copy16(const bf16* gsrc, bf16* ldst) {
#if defined(__AMDGCN__) && __has_builtin(__builtin_amdgcn_global_load_async_to_lds_b128)
  __builtin_amdgcn_global_load_async_to_lds_b128(
      (__attribute__((address_space(1))) v4i_g*)gsrc,
      (__attribute__((address_space(3))) v4i_g*)ldst, 0, 0);
#else
  *(v8bf*)ldst = *(const v8bf*)gsrc;   // register-bounce fallback
#endif
}

__device__ __forceinline__ void wait_async_all() {
#if defined(__AMDGCN__) && __has_builtin(__builtin_amdgcn_s_wait_asynccnt)
  __builtin_amdgcn_s_wait_asynccnt(0);
#elif defined(__AMDGCN__)
  asm volatile("s_wait_asynccnt 0" ::: "memory");
#endif
}

// ---------------------------------------------------------------------------
// fp32 -> bf16 conversion (weights)
// ---------------------------------------------------------------------------
__global__ void f32_to_bf16(const float* __restrict__ in, bf16* __restrict__ out,
                            size_t n) {
  size_t i = (size_t)blockIdx.x * blockDim.x + threadIdx.x;
  if (i < n) out[i] = (bf16)in[i];
}

// ---------------------------------------------------------------------------
// Fused LayerNorm (biased var, eps=1e-5) + RoPE (quirk: inv-freq exponent over
// FULL H=576, sliced to HD/2=32).  One block per token, 288 threads = 288 pairs.
// ---------------------------------------------------------------------------
__global__ __launch_bounds__(288)
void ln_rope_kernel(const float* __restrict__ x, const float* __restrict__ lnw,
                    const float* __restrict__ lnb, bf16* __restrict__ h) {
  const int tok = blockIdx.x;        // b*S + s
  const int s = tok & (Sv - 1);
  const int p = threadIdx.x;         // pair index 0..287
  const int head = p >> 5;           // /32
  const int j = p & 31;              // rotary pair within head
  const int e0 = head * HDv + 2 * j;
  const size_t base = (size_t)tok * Hv;
  const float x0 = x[base + e0];
  const float x1 = x[base + e0 + 1];

  float s1 = x0 + x1, s2 = x0 * x0 + x1 * x1;
  for (int off = 16; off >= 1; off >>= 1) {
    s1 += __shfl_xor(s1, off, 32);
    s2 += __shfl_xor(s2, off, 32);
  }
  __shared__ float r1[9], r2[9];
  const int wid = threadIdx.x >> 5;
  if ((threadIdx.x & 31) == 0) { r1[wid] = s1; r2[wid] = s2; }
  __syncthreads();
  float t1 = 0.f, t2 = 0.f;
#pragma unroll
  for (int w = 0; w < 9; ++w) { t1 += r1[w]; t2 += r2[w]; }
  const float mu = t1 * (1.0f / Hv);
  const float var = t2 * (1.0f / Hv) - mu * mu;
  const float rstd = rsqrtf(var + 1e-5f);

  float h0 = (x0 - mu) * rstd * lnw[e0] + lnb[e0];
  float h1 = (x1 - mu) * rstd * lnw[e0 + 1] + lnb[e0 + 1];

  // inv[j] = theta^(-(2j)/H)  (full-H exponent quirk)
  const float inv = __powf(10000.0f, -(2.0f * (float)j) * (1.0f / (float)Hv));
  const float ang = (float)s * inv;
  float sn, cs;
  __sincosf(ang, &sn, &cs);
  h[base + e0]     = (bf16)(h0 * cs - h1 * sn);
  h[base + e0 + 1] = (bf16)(h0 * sn + h1 * cs);
}

// ---------------------------------------------------------------------------
// Generic GEMM: C[M,N] = A[M,K] (bf16) @ W[N,K]^T (bf16), f32 accumulate.
// 256 threads = 8 waves; block tile 128x64; wave tile 16x64 (4 accumulators).
// Weight (B) tile 64x32 is shared by all 8 waves -> staged into LDS with
// async global->LDS copies, double-buffered across the K loop.
// Cf!=nullptr -> fp32 output (+ optional residual); else bf16 output.
// ---------------------------------------------------------------------------
__global__ __launch_bounds__(256)
void gemm_bf16_wmma(const bf16* __restrict__ A, const bf16* __restrict__ W,
                    bf16* __restrict__ Cb, float* __restrict__ Cf,
                    const float* __restrict__ resid, int M, int N, int K) {
  const int lane = threadIdx.x & 31;
  const int wave = threadIdx.x >> 5;   // 0..7
  const int half = lane >> 4;
  const int ln16 = lane & 15;
  const int m0 = blockIdx.x * 128 + wave * 16;
  const int n0 = blockIdx.y * 64;

  // LDS double buffer for the 64x32 bf16 weight tile (4 KB each).
  __shared__ __align__(16) bf16 wtile[2][64 * 32];

  // Each of the 256 threads stages one 16B chunk: row t/4, k-col (t%4)*8.
  const int srow = threadIdx.x >> 2;        // 0..63
  const int scol = (threadIdx.x & 3) * 8;   // 0,8,16,24
  const bf16* gstage = W + (size_t)(n0 + srow) * K + scol;

  v8f acc[4] = {};
  const bf16* arow = A + (size_t)(m0 + ln16) * K;

  const int nk = K / 32;
  async_copy16(gstage, &wtile[0][srow * 32 + scol]);   // prime buffer 0

  for (int i = 0; i < nk; ++i) {
    wait_async_all();        // tile i landed in LDS (this wave's copies)
    __syncthreads();         // ... and everyone else's
    if (i + 1 < nk)          // stream tile i+1 while computing on tile i
      async_copy16(gstage + (i + 1) * 32,
                   &wtile[(i + 1) & 1][srow * 32 + scol]);

    const int kk = i * 32;
    __builtin_prefetch(arow + kk + 64, 0, 3);          // A-stream prefetch
    v16bf a = load_frag16(arow + kk, half);
    const bf16* wt = wtile[i & 1];
#pragma unroll
    for (int nt = 0; nt < 4; ++nt) {
      v16bf b = load_frag16(wt + (nt * 16 + ln16) * 32, half);
      acc[nt] = __builtin_amdgcn_wmma_f32_16x16x32_bf16(
          false, a, false, b, (short)0, acc[nt], false, false);
    }
  }

#pragma unroll
  for (int nt = 0; nt < 4; ++nt) {
#pragma unroll
    for (int r = 0; r < 8; ++r) {
      const int row = m0 + r + 8 * half;       // C layout: M = r + 8*(lane/16)
      const int col = n0 + nt * 16 + ln16;     //           N = lane%16
      const size_t idx = (size_t)row * N + col;
      const float v = acc[nt][r];
      if (Cf) Cf[idx] = v + (resid ? resid[idx] : 0.0f);
      else    Cb[idx] = (bf16)v;
    }
  }
}

// ---------------------------------------------------------------------------
// Quirky reshape scatter: qkv flat [B,S,3H] == [B,NH,S,3HD].
// q,k -> [B,NH,S,HD] row-major;  v -> transposed [B,NH,HD,S] so the P@V
// B-fragments are contiguous loads.
// ---------------------------------------------------------------------------
__global__ void qkv_scatter(const bf16* __restrict__ qkv, bf16* __restrict__ q,
                            bf16* __restrict__ k, bf16* __restrict__ vT) {
  const size_t per_b = (size_t)Sv * (3 * Hv);
  size_t i = (size_t)blockIdx.x * blockDim.x + threadIdx.x;
  if (i >= (size_t)Bv * per_b) return;
  const int b = (int)(i / per_b);
  size_t off = i - (size_t)b * per_b;          // within-batch flat offset
  const int nh = (int)(off / ((size_t)Sv * 3 * HDv));
  size_t rem = off - (size_t)nh * Sv * 3 * HDv;
  const int s = (int)(rem / (3 * HDv));
  const int c = (int)(rem - (size_t)s * (3 * HDv));
  const bf16 v = qkv[i];
  const size_t ho = ((size_t)b * NHv + nh) * Sv;     // head-row offset (tokens)
  if (c < HDv)            q[(ho + s) * HDv + c] = v;
  else if (c < 2 * HDv)   k[(ho + s) * HDv + (c - HDv)] = v;
  else                    vT[ho * HDv + (size_t)(c - 2 * HDv) * Sv + s] = v;
}

// ---------------------------------------------------------------------------
// Flash attention, non-causal softmax over full S.  Grid (S/64, NH, B),
// 128 threads = 4 waves; each wave owns 16 query rows, loops 64-key chunks.
// Q@K^T and P@V are both WMMA; C-layout P goes through a per-wave LDS tile
// to be reloaded in A-fragment layout (DS ops are wave-ordered).
// ---------------------------------------------------------------------------
__global__ __launch_bounds__(128)
void attn_flash(const bf16* __restrict__ q, const bf16* __restrict__ k,
                const bf16* __restrict__ vT, bf16* __restrict__ o) {
  const int b = blockIdx.z, nh = blockIdx.y, qb = blockIdx.x;
  const int lane = threadIdx.x & 31, wave = threadIdx.x >> 5;
  const int half = lane >> 4, ln16 = lane & 15;
  const int m0 = qb * 64 + wave * 16;
  const size_t headoff = ((size_t)b * NHv + nh) * Sv;
  const bf16* Q  = q  + headoff * HDv;
  const bf16* Kp = k  + headoff * HDv;
  const bf16* Vt = vT + headoff * HDv;            // [HD][S]

  __shared__ __align__(16) bf16 p_lds[4][16 * 64];
  bf16* pl = p_lds[wave];

  const bf16* qrow = Q + (size_t)(m0 + ln16) * HDv;
  const v16bf aq0 = load_frag16(qrow + 0, half);
  const v16bf aq1 = load_frag16(qrow + 32, half);

  v8f oacc[4] = {};
  float mrow[8], lrow[8];
#pragma unroll
  for (int r = 0; r < 8; ++r) { mrow[r] = -1e30f; lrow[r] = 0.0f; }
  const float scale = 0.125f;                     // 1/sqrt(HD)

  for (int kc = 0; kc < Sv; kc += 64) {
    // ---- scores S = Q @ K^T for 16x64 tile -------------------------------
    v8f sfr[4] = {};
#pragma unroll
    for (int nt = 0; nt < 4; ++nt) {
      const bf16* krow = Kp + (size_t)(kc + nt * 16 + ln16) * HDv;
      v16bf bk0 = load_frag16(krow + 0, half);
      v16bf bk1 = load_frag16(krow + 32, half);
      sfr[nt] = __builtin_amdgcn_wmma_f32_16x16x32_bf16(
          false, aq0, false, bk0, (short)0, sfr[nt], false, false);
      sfr[nt] = __builtin_amdgcn_wmma_f32_16x16x32_bf16(
          false, aq1, false, bk1, (short)0, sfr[nt], false, false);
    }
    // ---- online softmax ---------------------------------------------------
    float nm[8], al[8];
#pragma unroll
    for (int r = 0; r < 8; ++r) {
      float mx = -1e30f;
#pragma unroll
      for (int nt = 0; nt < 4; ++nt) {
        sfr[nt][r] *= scale;
        mx = fmaxf(mx, sfr[nt][r]);
      }
      for (int off = 8; off >= 1; off >>= 1)     // reduce across half-wave
        mx = fmaxf(mx, __shfl_xor(mx, off, 16));
      nm[r] = fmaxf(mrow[r], mx);
      al[r] = __expf(mrow[r] - nm[r]);
      mrow[r] = nm[r];
    }
#pragma unroll
    for (int r = 0; r < 8; ++r) {
      float ps = 0.f;
#pragma unroll
      for (int nt = 0; nt < 4; ++nt) {
        float pv = __expf(sfr[nt][r] - nm[r]);
        ps += pv;
        pl[(r + 8 * half) * 64 + nt * 16 + ln16] = (bf16)pv;
      }
      for (int off = 8; off >= 1; off >>= 1)
        ps += __shfl_xor(ps, off, 16);
      lrow[r] = lrow[r] * al[r] + ps;
#pragma unroll
      for (int nt = 0; nt < 4; ++nt) oacc[nt][r] *= al[r];
    }
    // ---- reload P in A-fragment layout, then O += P @ V -------------------
    const bf16* prow = pl + ln16 * 64;
    v16bf ap0 = load_frag16(prow + 0, half);
    v16bf ap1 = load_frag16(prow + 32, half);
#pragma unroll
    for (int nt = 0; nt < 4; ++nt) {
      const bf16* vrow = Vt + (size_t)(nt * 16 + ln16) * Sv + kc;
      v16bf bv0 = load_frag16(vrow + 0, half);
      v16bf bv1 = load_frag16(vrow + 32, half);
      oacc[nt] = __builtin_amdgcn_wmma_f32_16x16x32_bf16(
          false, ap0, false, bv0, (short)0, oacc[nt], false, false);
      oacc[nt] = __builtin_amdgcn_wmma_f32_16x16x32_bf16(
          false, ap1, false, bv1, (short)0, oacc[nt], false, false);
    }
  }
  // ---- normalize + store; flat [B,NH,S,HD] == [B,S,H] for the next GEMM ---
  bf16* ob = o + (headoff + m0) * HDv;
#pragma unroll
  for (int r = 0; r < 8; ++r) {
    const float inv_l = 1.0f / lrow[r];
#pragma unroll
    for (int nt = 0; nt < 4; ++nt)
      ob[(size_t)(r + 8 * half) * HDv + nt * 16 + ln16] =
          (bf16)(oacc[nt][r] * inv_l);
  }
}

// ---------------------------------------------------------------------------
// SwiGLU: out[r,j] = silu(gu[r,j]) * gu[r, IM+j]
// ---------------------------------------------------------------------------
__global__ void swiglu_kernel(const bf16* __restrict__ gu, bf16* __restrict__ out) {
  size_t i = (size_t)blockIdx.x * blockDim.x + threadIdx.x;
  if (i >= (size_t)Mtok * IMv) return;
  const size_t r = i / IMv, j = i - r * IMv;
  const float g = (float)gu[r * (2 * IMv) + j];
  const float u = (float)gu[r * (2 * IMv) + IMv + j];
  out[i] = (bf16)((g / (1.0f + __expf(-g))) * u);
}

// ---------------------------------------------------------------------------
extern "C" void kernel_launch(void* const* d_in, const int* in_sizes, int n_in,
                              void* d_out, int out_size, void* d_ws, size_t ws_size,
                              hipStream_t stream) {
  const float* x    = (const float*)d_in[0];
  const float* lnw  = (const float*)d_in[1];
  const float* lnb  = (const float*)d_in[2];
  const float* wqkv = (const float*)d_in[3];
  const float* wo   = (const float*)d_in[4];
  const float* wgu  = (const float*)d_in[5];
  const float* wd   = (const float*)d_in[6];
  float* out = (float*)d_out;

  // ---- workspace carve (bf16 buffers, 256B aligned; ~168 MB total) --------
  char* p = (char*)d_ws;
  auto carve = [&p](size_t elems) {
    bf16* r = (bf16*)p;
    p += ((elems * sizeof(bf16) + 255) & ~(size_t)255);
    return r;
  };
  bf16* wq_bf   = carve((size_t)3 * Hv * Hv);        // [1728,576]
  bf16* wo_bf   = carve((size_t)Hv * Hv);            // [576,576]
  bf16* wgu_bf  = carve((size_t)2 * IMv * Hv);       // [3072,576]
  bf16* wd_bf   = carve((size_t)Hv * IMv);           // [576,1536]
  bf16* h_bf    = carve((size_t)Mtok * Hv);          // ln+rope output
  bf16* qkv_bf  = carve((size_t)Mtok * 3 * Hv);      // qkv proj output
  bf16* q_bf    = carve((size_t)Mtok * Hv);
  bf16* k_bf    = carve((size_t)Mtok * Hv);
  bf16* vT_bf   = carve((size_t)Mtok * Hv);
  bf16* attn_bf = carve((size_t)Mtok * Hv);          // attention output (flat)
  bf16* proj_bf = carve((size_t)Mtok * Hv);          // after w_o
  bf16* gu_bf   = carve((size_t)Mtok * 2 * IMv);     // gate_up output
  bf16* mlp_bf  = carve((size_t)Mtok * IMv);         // swiglu output
  (void)ws_size; (void)n_in; (void)in_sizes; (void)out_size;

  // ---- 0) weights fp32 -> bf16 --------------------------------------------
  auto cvt = [&](const float* src, bf16* dst, size_t n) {
    f32_to_bf16<<<dim3((unsigned)((n + 255) / 256)), dim3(256), 0, stream>>>(src, dst, n);
  };
  cvt(wqkv, wq_bf,  (size_t)3 * Hv * Hv);
  cvt(wo,   wo_bf,  (size_t)Hv * Hv);
  cvt(wgu,  wgu_bf, (size_t)2 * IMv * Hv);
  cvt(wd,   wd_bf,  (size_t)Hv * IMv);

  // ---- 1) LayerNorm + RoPE ------------------------------------------------
  ln_rope_kernel<<<dim3(Mtok), dim3(288), 0, stream>>>(x, lnw, lnb, h_bf);

  // ---- 2) QKV projection: [8192,576] x [1728,576]^T -----------------------
  gemm_bf16_wmma<<<dim3(Mtok / 128, (3 * Hv) / 64), dim3(256), 0, stream>>>(
      h_bf, wq_bf, qkv_bf, nullptr, nullptr, Mtok, 3 * Hv, Hv);

  // ---- 3) quirky-reshape scatter to q / k / v^T ---------------------------
  {
    size_t n = (size_t)Mtok * 3 * Hv;
    qkv_scatter<<<dim3((unsigned)((n + 255) / 256)), dim3(256), 0, stream>>>(
        qkv_bf, q_bf, k_bf, vT_bf);
  }

  // ---- 4) flash attention -------------------------------------------------
  attn_flash<<<dim3(Sv / 64, NHv, Bv), dim3(128), 0, stream>>>(
      q_bf, k_bf, vT_bf, attn_bf);

  // ---- 5) output projection: [8192,576] x [576,576]^T ---------------------
  gemm_bf16_wmma<<<dim3(Mtok / 128, Hv / 64), dim3(256), 0, stream>>>(
      attn_bf, wo_bf, proj_bf, nullptr, nullptr, Mtok, Hv, Hv);

  // ---- 6) gate_up: [8192,576] x [3072,576]^T ------------------------------
  gemm_bf16_wmma<<<dim3(Mtok / 128, (2 * IMv) / 64), dim3(256), 0, stream>>>(
      proj_bf, wgu_bf, gu_bf, nullptr, nullptr, Mtok, 2 * IMv, Hv);

  // ---- 7) SwiGLU ----------------------------------------------------------
  {
    size_t n = (size_t)Mtok * IMv;
    swiglu_kernel<<<dim3((unsigned)((n + 255) / 256)), dim3(256), 0, stream>>>(
        gu_bf, mlp_bf);
  }

  // ---- 8) down projection + residual -> fp32 d_out ------------------------
  gemm_bf16_wmma<<<dim3(Mtok / 128, Hv / 64), dim3(256), 0, stream>>>(
      mlp_bf, wd_bf, nullptr, out, x, Mtok, Hv, IMv);
}
